// SparseBiasedMHA_12584254177664
// MI455X (gfx1250) — compile-verified, matrix operand order checked by reading, not run on hardware
//
#include <hip/hip_runtime.h>

// ---------------------------------------------------------------------------
// SparseBiasedMHA for MI455X (gfx1250, wave32, WMMA 16x16x32 bf16,
// async global->LDS staging with padded LDS rows, WGP$-reuse for weights)
// ---------------------------------------------------------------------------

typedef __bf16 bf16_t;
typedef __attribute__((ext_vector_type(8)))  __bf16 v8bf;
typedef __attribute__((ext_vector_type(16))) __bf16 v16bf;
typedef __attribute__((ext_vector_type(8)))  float  v8f;

#define FEAT 256
#define HEADS 8
#define HEAD_DIM 32
#define SQRT_HD 5.656854249492381f  // sqrt(32) == 1/SCALING
#define ROWS_PB 128                 // rows per block in GEMMs (8 row-tiles)
#define XT_LD   (FEAT + 8)          // padded LDS row: 528B stride, bank-conflict-free

// ---- WMMA fragment helper (per ISA 7.12.2 16-bit layouts) -----------------
// A (16x32, MxK): lane m = lane&15; lanes 0-15 hold K = {0..7, 16..23},
// lanes 16-31 hold K = {8..15, 24..31}. B (32x16, KxN): symmetric -> lane
// holds column j = lane&15 = a row of W (for X @ W^T), same K split.
// Works for both global and LDS row pointers (addrspace inferred after inline).
__device__ __forceinline__ v16bf load_frag(const bf16_t* rowp, int k0, int lane) {
    int koff = k0 + ((lane < 16) ? 0 : 8);
    const bf16_t* p = rowp + koff;
    v8bf lo = *(const v8bf*)(p);
    v8bf hi = *(const v8bf*)(p + 16);
    v16bf r;
#pragma unroll
    for (int i = 0; i < 8; ++i) { r[i] = lo[i]; r[i + 8] = hi[i]; }
    return r;
}

// ---- float atomic max via int punning (hardware atomics, no CAS loop) -----
__device__ __forceinline__ void atomicMaxF(float* addr, float val) {
    if (!(val < 0.0f))
        atomicMax((int*)addr, __float_as_int(val));
    else
        atomicMin((unsigned int*)addr, __float_as_uint(val));
}

// ---------------------------------------------------------------------------
// 1) weight conversion: Wq (scaled by sqrt(HD)), Wk, Wv -> Wqkv_b; Wo -> Wo_b
// ---------------------------------------------------------------------------
__global__ void conv_weights_kernel(const float* __restrict__ Wq,
                                    const float* __restrict__ Wk,
                                    const float* __restrict__ Wv,
                                    const float* __restrict__ Wo,
                                    bf16_t* __restrict__ Wqkv_b,
                                    bf16_t* __restrict__ Wo_b) {
    int gid = blockIdx.x * blockDim.x + threadIdx.x;
    if (gid >= 4 * FEAT * FEAT) return;
    int w = gid >> 16;
    int idx = gid & 0xFFFF;
    float val;
    if (w == 0)      val = Wq[idx] * SQRT_HD;
    else if (w == 1) val = Wk[idx];
    else if (w == 2) val = Wv[idx];
    else             val = Wo[idx];
    if (w < 3) Wqkv_b[(size_t)w * FEAT * FEAT + idx] = (bf16_t)val;
    else       Wo_b[idx] = (bf16_t)val;
}

// ---------------------------------------------------------------------------
// 2) nfeat f32 -> bf16
// ---------------------------------------------------------------------------
__global__ void conv_x_kernel(const float* __restrict__ X, bf16_t* __restrict__ Xb,
                              int total) {
    int gid = blockIdx.x * blockDim.x + threadIdx.x;
    if (gid < total) Xb[gid] = (bf16_t)X[gid];
}

// ---------------------------------------------------------------------------
// 3) init: m=-inf, s=0, outacc=0   (required every launch; atomics accumulate)
// ---------------------------------------------------------------------------
__global__ void init_kernel(float* __restrict__ mmax, float* __restrict__ ssum,
                            float* __restrict__ outacc, int N) {
    int gid = blockIdx.x * blockDim.x + threadIdx.x;
    if (gid < N * HEADS) { mmax[gid] = -__builtin_inff(); ssum[gid] = 0.0f; }
    if (gid < N * FEAT)  outacc[gid] = 0.0f;
}

// ---------------------------------------------------------------------------
// 4) fused QKV projection: Q|K|V = X @ W{q,k,v}^T  (bf16 WMMA)
//    block = 128 rows; X async-staged to padded LDS; weights streamed from
//    global with 8x WGP$ reuse (t outer, r inner); no fragment residency.
// ---------------------------------------------------------------------------
__global__ __launch_bounds__(256) void qkv_gemm_kernel(
        const bf16_t* __restrict__ Xb, const bf16_t* __restrict__ Wqkv_b,
        bf16_t* __restrict__ Qb, bf16_t* __restrict__ Kb, bf16_t* __restrict__ Vb,
        int N) {
    __shared__ __align__(16) bf16_t xtile[ROWS_PB * XT_LD];  // ~66 KB

    const int lane = threadIdx.x & 31;
    const int wave = threadIdx.x >> 5;
    const int row0 = blockIdx.x * ROWS_PB;

    // ---- async stage X[row0 .. row0+127][:] into LDS (16B chunks) ----
    {
        const int nchunks = ROWS_PB * FEAT / 8;  // 4096 chunks of 8 bf16 (16B)
        for (int ch = threadIdx.x; ch < nchunks; ch += 256) {
            int row  = ch >> 5;        // 32 chunks per 512B row
            int col8 = ch & 31;
            int grow = row0 + row; if (grow >= N) grow = N - 1;
            const bf16_t* gsrc = Xb + (size_t)grow * FEAT + col8 * 8;
            unsigned lds_off = (unsigned)(size_t)&xtile[row * XT_LD + col8 * 8];
            asm volatile("global_load_async_to_lds_b128 %0, %1, off"
                         :: "v"(lds_off), "v"(gsrc) : "memory");
        }
        asm volatile("s_wait_asynccnt 0x0" ::: "memory");
        __syncthreads();
    }

    const int n     = lane & 15;
    const int mbase = (lane >> 4) * 8;
    const bool full = (row0 + ROWS_PB <= N);   // block-uniform store fast path

#pragma unroll 1
    for (int t = wave; t < 48; t += 8) {          // 3 matrices * 16 col tiles
        const int mat = t >> 4;
        const int j0  = (t & 15) * 16;
        const bf16_t* wrow = Wqkv_b + (size_t)mat * FEAT * FEAT + (size_t)(j0 + n) * FEAT;
        bf16_t* O = (mat == 0) ? Qb : (mat == 1) ? Kb : Vb;

#pragma unroll 1
        for (int r = 0; r < ROWS_PB / 16; ++r) {  // 8 row tiles; W re-read hits WGP$
            const bf16_t* arow = &xtile[(r * 16 + n) * XT_LD];
            v8f acc = {};
#pragma unroll
            for (int kk = 0; kk < 8; ++kk) {
                v16bf afrag = load_frag(arow, kk * 32, lane);   // ds_load
                v16bf bfrag = load_frag(wrow, kk * 32, lane);   // global_load
                acc = __builtin_amdgcn_wmma_f32_16x16x32_bf16(
                    false, afrag, false, bfrag, (short)0, acc, false, false);
            }
            bf16_t* obase = O + (size_t)(row0 + r * 16 + mbase) * FEAT + j0 + n;
            if (full) {
#pragma unroll
                for (int rr = 0; rr < 8; ++rr)
                    obase[(size_t)rr * FEAT] = (bf16_t)acc[rr];
            } else {
#pragma unroll
                for (int rr = 0; rr < 8; ++rr) {
                    int orow = row0 + r * 16 + mbase + rr;
                    if (orow < N) O[(size_t)orow * FEAT + j0 + n] = (bf16_t)acc[rr];
                }
            }
        }
    }
}

// ---------------------------------------------------------------------------
// 5) per-edge logits: attn[e,h] = <k[src],q[dst]>_h + bias[e,h]; seg-max(dst)
// ---------------------------------------------------------------------------
__global__ void edge_logits_kernel(const bf16_t* __restrict__ Kb,
                                   const bf16_t* __restrict__ Qb,
                                   const float* __restrict__ bias,
                                   const int* __restrict__ src,
                                   const int* __restrict__ dst,
                                   float* __restrict__ logits,
                                   float* __restrict__ mmax, int E) {
    int gid = blockIdx.x * blockDim.x + threadIdx.x;
    if (gid >= E * HEADS) return;
    const int e = gid >> 3, h = gid & 7;
    const int s = src[e], d = dst[e];
    const v8bf* kp = (const v8bf*)(Kb + (size_t)s * FEAT + h * HEAD_DIM);
    const v8bf* qp = (const v8bf*)(Qb + (size_t)d * FEAT + h * HEAD_DIM);
    float acc = 0.0f;
#pragma unroll
    for (int c = 0; c < 4; ++c) {
        v8bf kv = kp[c], qv = qp[c];
#pragma unroll
        for (int i = 0; i < 8; ++i) acc += (float)kv[i] * (float)qv[i];
    }
    acc += bias[gid];
    logits[gid] = acc;
    atomicMaxF(&mmax[d * HEADS + h], acc);
}

// ---------------------------------------------------------------------------
// 6) a = exp(attn - m[dst]); segment-sum into s[dst]
// ---------------------------------------------------------------------------
__global__ void edge_exp_kernel(const int* __restrict__ dst,
                                const float* __restrict__ mmax,
                                float* __restrict__ logits,
                                float* __restrict__ ssum, int E) {
    int gid = blockIdx.x * blockDim.x + threadIdx.x;
    if (gid >= E * HEADS) return;
    const int e = gid >> 3, h = gid & 7;
    const int d = dst[e];
    float a = __expf(logits[gid] - mmax[d * HEADS + h]);
    logits[gid] = a;
    atomicAdd(&ssum[d * HEADS + h], a);
}

// ---------------------------------------------------------------------------
// 7) normalize + scatter: outacc[dst,h,:] += (a/s[dst,h]) * v[src,h,:]
// ---------------------------------------------------------------------------
__global__ void edge_scatter_kernel(const int* __restrict__ src,
                                    const int* __restrict__ dst,
                                    const float* __restrict__ logits,
                                    const float* __restrict__ ssum,
                                    const bf16_t* __restrict__ Vb,
                                    float* __restrict__ outacc, int E) {
    int gid = blockIdx.x * blockDim.x + threadIdx.x;
    if (gid >= E * HEADS) return;
    const int e = gid >> 3, h = gid & 7;
    const int s = src[e], d = dst[e];
    const float w = logits[gid] / ssum[d * HEADS + h];
    const v8bf* vp = (const v8bf*)(Vb + (size_t)s * FEAT + h * HEAD_DIM);
    float* op = outacc + (size_t)d * FEAT + h * HEAD_DIM;
#pragma unroll
    for (int c = 0; c < 4; ++c) {
        v8bf vv = vp[c];
#pragma unroll
        for (int i = 0; i < 8; ++i) atomicAdd(op + c * 8 + i, w * (float)vv[i]);
    }
}

// ---------------------------------------------------------------------------
// 8) output projection: Out = outacc @ Wo^T
//    block = 128 rows; f32 -> bf16 conversion during LDS staging (padded);
//    t outer / r inner with WGP$ weight reuse, no fragment residency.
// ---------------------------------------------------------------------------
__global__ __launch_bounds__(256) void out_gemm_kernel(
        const float* __restrict__ Acc, const bf16_t* __restrict__ Wo_b,
        float* __restrict__ Out, int N) {
    __shared__ __align__(16) bf16_t xtile[ROWS_PB * XT_LD];  // ~66 KB

    const int lane = threadIdx.x & 31;
    const int wave = threadIdx.x >> 5;
    const int row0 = blockIdx.x * ROWS_PB;

    // ---- stage + convert f32 -> bf16 into padded LDS ----
    {
        const int nchunks = ROWS_PB * FEAT / 8;  // 4096 chunks of 8 elements
        for (int ch = threadIdx.x; ch < nchunks; ch += 256) {
            int row  = ch >> 5;
            int col8 = ch & 31;
            int grow = row0 + row; if (grow >= N) grow = N - 1;
            const float* g = Acc + (size_t)grow * FEAT + col8 * 8;
            v8bf tmp;
#pragma unroll
            for (int i = 0; i < 8; ++i) tmp[i] = (bf16_t)g[i];
            *(v8bf*)&xtile[row * XT_LD + col8 * 8] = tmp;
        }
        __syncthreads();
    }

    const int n     = lane & 15;
    const int mbase = (lane >> 4) * 8;
    const bool full = (row0 + ROWS_PB <= N);

#pragma unroll 1
    for (int t = wave; t < 16; t += 8) {          // 16 col tiles / 8 waves
        const int j0 = t * 16;
        const bf16_t* wrow = Wo_b + (size_t)(j0 + n) * FEAT;

#pragma unroll 1
        for (int r = 0; r < ROWS_PB / 16; ++r) {
            const bf16_t* arow = &xtile[(r * 16 + n) * XT_LD];
            v8f acc = {};
#pragma unroll
            for (int kk = 0; kk < 8; ++kk) {
                v16bf afrag = load_frag(arow, kk * 32, lane);
                v16bf bfrag = load_frag(wrow, kk * 32, lane);
                acc = __builtin_amdgcn_wmma_f32_16x16x32_bf16(
                    false, afrag, false, bfrag, (short)0, acc, false, false);
            }
            float* obase = Out + (size_t)(row0 + r * 16 + mbase) * FEAT + j0 + n;
            if (full) {
#pragma unroll
                for (int rr = 0; rr < 8; ++rr)
                    obase[(size_t)rr * FEAT] = acc[rr];
            } else {
#pragma unroll
                for (int rr = 0; rr < 8; ++rr) {
                    int orow = row0 + r * 16 + mbase + rr;
                    if (orow < N) Out[(size_t)orow * FEAT + j0 + n] = acc[rr];
                }
            }
        }
    }
}

// ---------------------------------------------------------------------------
extern "C" void kernel_launch(void* const* d_in, const int* in_sizes, int n_in,
                              void* d_out, int out_size, void* d_ws, size_t ws_size,
                              hipStream_t stream) {
    const float* nfeat = (const float*)d_in[0];
    const float* bias  = (const float*)d_in[1];
    const float* Wq    = (const float*)d_in[2];
    const float* Wk    = (const float*)d_in[3];
    const float* Wv    = (const float*)d_in[4];
    const float* Wo    = (const float*)d_in[5];
    const int*   src   = (const int*)d_in[6];
    const int*   dst   = (const int*)d_in[7];
    float* out = (float*)d_out;

    const int N = in_sizes[0] / FEAT;
    const int E = in_sizes[1] / HEADS;

    char*  w   = (char*)d_ws;
    size_t off = 0;
    auto carve = [&](size_t bytes) -> char* {
        char* p = w + off;
        off = (off + bytes + 255) & ~(size_t)255;
        return p;
    };
    bf16_t* Xb     = (bf16_t*)carve((size_t)N * FEAT * 2);
    bf16_t* Qb     = (bf16_t*)carve((size_t)N * FEAT * 2);
    bf16_t* Kb     = (bf16_t*)carve((size_t)N * FEAT * 2);
    bf16_t* Vb     = (bf16_t*)carve((size_t)N * FEAT * 2);
    bf16_t* Wqkv_b = (bf16_t*)carve((size_t)3 * FEAT * FEAT * 2);
    bf16_t* Wo_b   = (bf16_t*)carve((size_t)FEAT * FEAT * 2);
    float*  logits = (float*)carve((size_t)E * HEADS * 4);
    float*  mmax   = (float*)carve((size_t)N * HEADS * 4);
    float*  ssum   = (float*)carve((size_t)N * HEADS * 4);
    float*  outacc = (float*)carve((size_t)N * FEAT * 4);
    (void)ws_size; (void)n_in; (void)out_size;

    const int TB = 256;
    const int gemm_blocks = (N + ROWS_PB - 1) / ROWS_PB;

    conv_weights_kernel<<<(4 * FEAT * FEAT + TB - 1) / TB, TB, 0, stream>>>(
        Wq, Wk, Wv, Wo, Wqkv_b, Wo_b);

    conv_x_kernel<<<((N * FEAT) + TB - 1) / TB, TB, 0, stream>>>(nfeat, Xb, N * FEAT);

    init_kernel<<<((N * FEAT) + TB - 1) / TB, TB, 0, stream>>>(mmax, ssum, outacc, N);

    qkv_gemm_kernel<<<gemm_blocks, TB, 0, stream>>>(Xb, Wqkv_b, Qb, Kb, Vb, N);

    const int EH = E * HEADS;
    edge_logits_kernel<<<(EH + TB - 1) / TB, TB, 0, stream>>>(
        Kb, Qb, bias, src, dst, logits, mmax, E);

    edge_exp_kernel<<<(EH + TB - 1) / TB, TB, 0, stream>>>(dst, mmax, logits, ssum, E);

    edge_scatter_kernel<<<(EH + TB - 1) / TB, TB, 0, stream>>>(
        src, dst, logits, ssum, Vb, outacc, E);

    out_gemm_kernel<<<gemm_blocks, TB, 0, stream>>>(outacc, Wo_b, out, N);
}
